// AttentionBlock_59820304498690
// MI455X (gfx1250) — matrix operand order repaired and checked by reading, not compiled, not run on hardware
//
#include <hip/hip_runtime.h>

#define DEVI __device__ __forceinline__

typedef __bf16 bf16;
typedef __bf16 bf16x4 __attribute__((ext_vector_type(4)));
typedef __bf16 bf16x8 __attribute__((ext_vector_type(8)));
typedef __bf16 v16bf  __attribute__((ext_vector_type(16)));
typedef float  v8f    __attribute__((ext_vector_type(8)));

static constexpr int  Bsz = 8, Cch = 128, Npos = 4096;
static constexpr int  CG  = 16;            // channels per group
static constexpr float EPSv = 1e-5f;
static constexpr float SCL  = 0.08838834764831845f; // 128^-0.5

DEVI v8f v8zero() { v8f r;
#pragma unroll
  for (int i = 0; i < 8; ++i) r[i] = 0.f;
  return r; }

DEVI v16bf frag_cat(bf16x8 a, bf16x8 b) { v16bf r;
#pragma unroll
  for (int i = 0; i < 8; ++i) { r[i] = a[i]; r[i + 8] = b[i]; }
  return r; }

// A-operand fragment from an LDS row holding 32 contiguous K values.
// Contiguous layout (16-bit A, 16x32): lanes 0-15 hold K={0..15}, lanes 16-31 K={16..31}
DEVI v16bf lds_fragA(const bf16* row, int hi) {
  const bf16* p0 = row + (hi ? 16 : 0);
  return frag_cat(*(const bf16x8*)p0, *(const bf16x8*)(p0 + 8));
}
// B-operand fragment: 16 contiguous K values starting at p (caller adds hi*16).
DEVI v16bf lds_fragB(const bf16* p) {
  return frag_cat(*(const bf16x8*)p, *(const bf16x8*)(p + 8));
}
DEVI v8f wmma_bf16(v16bf a, v16bf b, v8f c) {
  return __builtin_amdgcn_wmma_f32_16x16x32_bf16(false, a, false, b, (short)0, c,
                                                 false, false);
}
DEVI float xormax16(float v) {
  v = fmaxf(v, __shfl_xor(v, 1));  v = fmaxf(v, __shfl_xor(v, 2));
  v = fmaxf(v, __shfl_xor(v, 4));  v = fmaxf(v, __shfl_xor(v, 8));
  return v;
}
DEVI float xorsum16(float v) {
  v += __shfl_xor(v, 1); v += __shfl_xor(v, 2);
  v += __shfl_xor(v, 4); v += __shfl_xor(v, 8);
  return v;
}

// ---------------- GroupNorm: per-(b,g) mean / rstd ----------------
__global__ void gn_stats(const float* __restrict__ x, float* __restrict__ stats) {
  const int bg = blockIdx.x;                       // 0..63 (b*8+g)
  const float* p = x + (size_t)bg * (CG * Npos);   // 16 contiguous channels
  float s = 0.f, ss = 0.f;
  for (int i = threadIdx.x; i < CG * Npos / 4; i += 256) {
    float4 v = *(const float4*)(p + (size_t)i * 4);
    s  += v.x + v.y + v.z + v.w;
    ss += v.x * v.x + v.y * v.y + v.z * v.z + v.w * v.w;
  }
  __shared__ float rs[256], rq[256];
  rs[threadIdx.x] = s; rq[threadIdx.x] = ss;
  __syncthreads();
  for (int o = 128; o > 0; o >>= 1) {
    if (threadIdx.x < o) { rs[threadIdx.x] += rs[threadIdx.x + o];
                           rq[threadIdx.x] += rq[threadIdx.x + o]; }
    __syncthreads();
  }
  if (threadIdx.x == 0) {
    float m   = rs[0] / (float)(CG * Npos);
    float var = rq[0] / (float)(CG * Npos) - m * m;
    stats[bg * 2]     = m;
    stats[bg * 2 + 1] = rsqrtf(var + EPSv);
  }
}

// ---------------- normalize + affine -> bf16 hn ----------------
__global__ void gn_norm(const float* __restrict__ x, const float* __restrict__ gamma,
                        const float* __restrict__ beta, const float* __restrict__ stats,
                        bf16* __restrict__ hn) {
  const size_t idx = ((size_t)blockIdx.x * 256 + threadIdx.x) * 4;
  const int c  = (int)((idx >> 12) & 127);
  const int b  = (int)(idx >> 19);
  const int bg = b * 8 + (c >> 4);
  const float m = stats[bg * 2], r = stats[bg * 2 + 1];
  const float ga = gamma[c] * r;
  const float be = beta[c] - m * ga;
  const float4 v = *(const float4*)(x + idx);
  bf16x4 o;
  o[0] = (bf16)(v.x * ga + be);
  o[1] = (bf16)(v.y * ga + be);
  o[2] = (bf16)(v.z * ga + be);
  o[3] = (bf16)(v.w * ga + be);
  *(bf16x4*)(hn + idx) = o;
}

// ---------------- 1x1 conv as GEMM: Out[b,o,n] = sum_c W[o,c]*Bin[b,c,n] + bias ----
// block tile: 128(M) x 128(N), K=128 in 4 chunks of 32. 8 waves, each 32x64.
template <bool OUT_BF16>
__global__ void __launch_bounds__(256, 1)
gemm_cn(const float* __restrict__ W, const float* __restrict__ bias,
        const bf16* __restrict__ Bin, void* __restrict__ Out,
        const float* __restrict__ Resid, int Mtot) {
  const int b = blockIdx.z, m0 = blockIdx.y * 128, n0 = blockIdx.x * 128;
  const int tid = threadIdx.x, lane = tid & 31, wv = tid >> 5;
  const int l15 = lane & 15, hi = lane >> 4;
  const int wm = wv & 3, wn = wv >> 2;          // 4x32 rows, 2x64 cols
  __shared__ __attribute__((aligned(16))) bf16 Asm[128 * 32];  // [o][k]
  __shared__ __attribute__((aligned(16))) bf16 Bsm[128 * 32];  // [n][k]

  v8f acc[2][4];
#pragma unroll
  for (int mo = 0; mo < 2; ++mo)
#pragma unroll
    for (int nn = 0; nn < 4; ++nn) acc[mo][nn] = v8zero();

  for (int kc = 0; kc < 4; ++kc) {
    __syncthreads();
#pragma unroll
    for (int t = 0; t < 4; ++t) {                // stage A: float4 load -> bf16x4
      int idx = tid + t * 256; int o = idx >> 3, kk = (idx & 7) * 4;
      float4 w4 = *(const float4*)&W[(size_t)(m0 + o) * 128 + kc * 32 + kk];
      bf16x4 a4;
      a4[0] = (bf16)w4.x; a4[1] = (bf16)w4.y; a4[2] = (bf16)w4.z; a4[3] = (bf16)w4.w;
      *(bf16x4*)&Asm[o * 32 + kk] = a4;
    }
#pragma unroll
    for (int t = 0; t < 4; ++t) {                // stage B: [k][n] -> [n][k]
      int idx = tid + t * 256; int kk = idx >> 5, n4 = (idx & 31) * 4;
      bf16x4 x4 = *(const bf16x4*)
          &Bin[((size_t)b * 128 + kc * 32 + kk) * (size_t)Npos + n0 + n4];
      Bsm[(n4 + 0) * 32 + kk] = x4[0];
      Bsm[(n4 + 1) * 32 + kk] = x4[1];
      Bsm[(n4 + 2) * 32 + kk] = x4[2];
      Bsm[(n4 + 3) * 32 + kk] = x4[3];
    }
    __syncthreads();
    v16bf af[2], bfr[4];
#pragma unroll
    for (int mo = 0; mo < 2; ++mo)
      af[mo] = lds_fragA(&Asm[(wm * 32 + mo * 16 + l15) * 32], hi);
#pragma unroll
    for (int nn = 0; nn < 4; ++nn)
      bfr[nn] = lds_fragB(&Bsm[(wn * 64 + nn * 16 + l15) * 32 + hi * 16]);
#pragma unroll
    for (int mo = 0; mo < 2; ++mo)
#pragma unroll
      for (int nn = 0; nn < 4; ++nn)
        acc[mo][nn] = wmma_bf16(af[mo], bfr[nn], acc[mo][nn]);
  }

#pragma unroll
  for (int mo = 0; mo < 2; ++mo)
#pragma unroll
    for (int nn = 0; nn < 4; ++nn)
#pragma unroll
      for (int r = 0; r < 8; ++r) {
        int o = m0 + wm * 32 + mo * 16 + r + hi * 8;
        int n = n0 + wn * 64 + nn * 16 + l15;
        float val = acc[mo][nn][r] + bias[o];
        size_t oi = ((size_t)b * Mtot + o) * (size_t)Npos + n;
        if (OUT_BF16) ((bf16*)Out)[oi] = (bf16)val;
        else          ((float*)Out)[oi] = val + Resid[oi];
      }
}

// ---------------- flash attention ----------------
// Block: 128 queries (8 waves x 16). Loop 64-key tiles, online softmax in VGPRs.
__global__ void __launch_bounds__(256, 1)
flash_attn(const bf16* __restrict__ qkv, bf16* __restrict__ outp) {
  const int b = blockIdx.y, i0 = blockIdx.x * 128;
  const int tid = threadIdx.x, lane = tid & 31, wv = tid >> 5;
  const int l15 = lane & 15, hi = lane >> 4;
  const bf16* qg = qkv + (size_t)b * 384 * Npos;
  const bf16* kg = qg + (size_t)128 * Npos;
  const bf16* vg = qg + (size_t)256 * Npos;
  __shared__ __attribute__((aligned(16))) bf16 Ksm[64 * 128];    // [j][c]
  __shared__ __attribute__((aligned(16))) bf16 Vsm[128 * 64];    // [c][j]
  __shared__ __attribute__((aligned(16))) bf16 Psm[8 * 16 * 64]; // per-wave [i][j]
  bf16* Pw = &Psm[wv * 16 * 64];

  // Q fragments resident in registers: A-layout gather (i fixed per lane, c runs)
  v16bf aq[4];
  {
    const int iq = i0 + wv * 16 + l15;
#pragma unroll
    for (int kc = 0; kc < 4; ++kc)
#pragma unroll
      for (int e = 0; e < 16; ++e) {
        int c = kc * 32 + hi * 16 + e;
        aq[kc][e] = qg[(size_t)c * Npos + iq];
      }
  }

  float m_i[8], l_i[8]; v8f acc[8];
#pragma unroll
  for (int r = 0; r < 8; ++r) { m_i[r] = -3.0e38f; l_i[r] = 0.f; }
#pragma unroll
  for (int cs = 0; cs < 8; ++cs) acc[cs] = v8zero();

  for (int jt = 0; jt < 64; ++jt) {
    const int j0 = jt * 64;
    __syncthreads();
#pragma unroll
    for (int t = 0; t < 8; ++t) {             // stage K (transposed) and V, 64-bit loads
      int idx = tid + t * 256; int c = idx >> 4, j4 = (idx & 15) * 4;
      bf16x4 k4 = *(const bf16x4*)&kg[(size_t)c * Npos + j0 + j4];
      bf16x4 v4 = *(const bf16x4*)&vg[(size_t)c * Npos + j0 + j4];
      *(bf16x4*)&Vsm[c * 64 + j4] = v4;
      Ksm[(j4 + 0) * 128 + c] = k4[0];
      Ksm[(j4 + 1) * 128 + c] = k4[1];
      Ksm[(j4 + 2) * 128 + c] = k4[2];
      Ksm[(j4 + 3) * 128 + c] = k4[3];
    }
    __syncthreads();
    if (jt < 63) {                            // gfx1250 global_prefetch of next tile
      __builtin_prefetch(&kg[j0 + 64], 0, 1);
      __builtin_prefetch(&vg[j0 + 64], 0, 1);
    }

    // S = Q^T K : 4 key-subtiles x 4 channel-chunks
    v8f s[4];
#pragma unroll
    for (int js = 0; js < 4; ++js) {
      v16bf bfr[4];
#pragma unroll
      for (int kc = 0; kc < 4; ++kc)
        bfr[kc] = lds_fragB(&Ksm[(js * 16 + l15) * 128 + kc * 32 + hi * 16]);
      v8f sv = v8zero();
#pragma unroll
      for (int kc = 0; kc < 4; ++kc) sv = wmma_bf16(aq[kc], bfr[kc], sv);
      s[js] = sv;
    }

    // online softmax (per row = VGPR slot r within each 16-lane half)
    float alpha[8];
#pragma unroll
    for (int r = 0; r < 8; ++r) {
      float mx = -3.0e38f;
#pragma unroll
      for (int js = 0; js < 4; ++js) { float t0 = s[js][r] * SCL; s[js][r] = t0; mx = fmaxf(mx, t0); }
      mx = xormax16(mx);
      float mn = fmaxf(m_i[r], mx);
      alpha[r] = __expf(m_i[r] - mn);
      float sum = 0.f;
#pragma unroll
      for (int js = 0; js < 4; ++js) { float p = __expf(s[js][r] - mn); s[js][r] = p; sum += p; }
      sum = xorsum16(sum);
      l_i[r] = l_i[r] * alpha[r] + sum;
      m_i[r] = mn;
    }

    // stage P (D-layout -> [i][j] rows in per-wave LDS; same-wave DS is in-order)
#pragma unroll
    for (int js = 0; js < 4; ++js)
#pragma unroll
      for (int r = 0; r < 8; ++r)
        Pw[(r + hi * 8) * 64 + js * 16 + l15] = (bf16)s[js][r];

    // rescale accumulators, then O += P x V^T (one P fragment live at a time)
#pragma unroll
    for (int cs = 0; cs < 8; ++cs)
#pragma unroll
      for (int r = 0; r < 8; ++r) acc[cs][r] *= alpha[r];

#pragma unroll
    for (int jc = 0; jc < 2; ++jc) {
      v16bf pf = lds_fragA(&Pw[l15 * 64 + jc * 32], hi);
#pragma unroll
      for (int cs = 0; cs < 8; ++cs)
        acc[cs] = wmma_bf16(pf,
                            lds_fragB(&Vsm[(cs * 16 + l15) * 64 + jc * 32 + hi * 16]),
                            acc[cs]);
    }
  }

#pragma unroll
  for (int r = 0; r < 8; ++r) l_i[r] = 1.f / l_i[r];
#pragma unroll
  for (int cs = 0; cs < 8; ++cs)
#pragma unroll
    for (int r = 0; r < 8; ++r) {
      int c = cs * 16 + l15;
      int i = i0 + wv * 16 + r + hi * 8;
      outp[((size_t)b * 128 + c) * (size_t)Npos + i] = (bf16)(acc[cs][r] * l_i[r]);
    }
}

extern "C" void kernel_launch(void* const* d_in, const int* in_sizes, int n_in,
                              void* d_out, int out_size, void* d_ws, size_t ws_size,
                              hipStream_t stream) {
  const float* x      = (const float*)d_in[0];
  const float* gamma  = (const float*)d_in[1];
  const float* beta   = (const float*)d_in[2];
  const float* w_qkv  = (const float*)d_in[3];
  const float* b_qkv  = (const float*)d_in[4];
  const float* w_proj = (const float*)d_in[5];
  const float* b_proj = (const float*)d_in[6];

  float* stats = (float*)d_ws;                                   // 64*2 floats
  bf16*  hn    = (bf16*)((char*)d_ws + 4096);                    // [B,128,N] bf16 (8MB)
  bf16*  qkvb  = (bf16*)((char*)d_ws + 4096 +
                         (size_t)Bsz * Cch * Npos * sizeof(bf16)); // [B,384,N] bf16 (24MB)

  gn_stats<<<64, 256, 0, stream>>>(x, stats);
  gn_norm<<<4096, 256, 0, stream>>>(x, gamma, beta, stats, hn);
  gemm_cn<true ><<<dim3(32, 3, 8), 256, 0, stream>>>(w_qkv, b_qkv, hn, (void*)qkvb,
                                                     nullptr, 384);
  // hn is fully consumed by the QKV GEMM above -> reuse its buffer as attn output
  flash_attn<<<dim3(32, 8), 256, 0, stream>>>(qkvb, hn);
  gemm_cn<false><<<dim3(32, 1, 8), 256, 0, stream>>>(w_proj, b_proj, hn, d_out,
                                                     x, 128);
}
